// CapsuleMortalityHead_89593017794951
// MI455X (gfx1250) — matrix operand order repaired and checked by reading, not compiled, not run on hardware
//
#include <hip/hip_runtime.h>

#define BTOT 8192
#define NCAP 7
#define DDIM 256
#define MCAP 25
#define VDIM 64
#define MV   1600   // MCAP*VDIM

// ---------------- workspace layout (bytes) ----------------
#define VOTES_BYTES ((size_t)BTOT * NCAP * MV * 2)          // 183,500,800 (bf16, < 192MB L2)
#define WSPL_ELEMS  ((size_t)NCAP * MV * DDIM)              // 2,867,200
#define ASPL_ELEMS  ((size_t)NCAP * BTOT * DDIM)            // 14,680,064
#define WHI_OFF     (VOTES_BYTES)
#define WLO_OFF     (WHI_OFF + WSPL_ELEMS * 2)
#define AHI_OFF     (WLO_OFF + WSPL_ELEMS * 2)
#define ALO_OFF     (AHI_OFF + ASPL_ELEMS * 2)

typedef __attribute__((ext_vector_type(16))) __bf16 v16bf;
typedef __attribute__((ext_vector_type(8)))  float  v8f;
typedef int v4i_t __attribute__((vector_size(16)));

struct __align__(16) U4 { unsigned x, y, z, w; };

__device__ __forceinline__ unsigned short f2bf(float f) {
  unsigned u = __float_as_uint(f);
  u += 0x7FFFu + ((u >> 16) & 1u);     // RNE to bf16
  return (unsigned short)(u >> 16);
}
__device__ __forceinline__ float bf2f(unsigned short h) {
  return __uint_as_float(((unsigned)h) << 16);
}
__device__ __forceinline__ float lo16f(unsigned u) { return __uint_as_float(u << 16); }
__device__ __forceinline__ float hi16f(unsigned u) { return __uint_as_float(u & 0xffff0000u); }
__device__ __forceinline__ float shfl_red8(float s) {   // reduce within 8-lane group
  s += __shfl_xor(s, 1, 32);
  s += __shfl_xor(s, 2, 32);
  s += __shfl_xor(s, 4, 32);
  return s;
}
__device__ __forceinline__ void pack8(unsigned short* dst, const unsigned short h[8]) {
  U4 r;
  r.x = (unsigned)h[0] | ((unsigned)h[1] << 16);
  r.y = (unsigned)h[2] | ((unsigned)h[3] << 16);
  r.z = (unsigned)h[4] | ((unsigned)h[5] << 16);
  r.w = (unsigned)h[6] | ((unsigned)h[7] << 16);
  *(U4*)dst = r;
}

// ---------------- CDNA5 async global->LDS copy (16B per lane) ----------------
#if defined(__gfx1250__) && __has_builtin(__builtin_amdgcn_global_load_async_to_lds_b128)
#define HAVE_ASYNC 1
#else
#define HAVE_ASYNC 0
#endif

__device__ __forceinline__ void cp16(const void* g, void* l) {
#if HAVE_ASYNC
  __builtin_amdgcn_global_load_async_to_lds_b128(
      (__attribute__((address_space(1))) v4i_t*)g,
      (__attribute__((address_space(3))) v4i_t*)l, 0, 0);
#else
  *(U4*)l = *(const U4*)g;
#endif
}
__device__ __forceinline__ void wait_async0() {
#if HAVE_ASYNC
#if __has_builtin(__builtin_amdgcn_s_wait_asynccnt)
  __builtin_amdgcn_s_wait_asynccnt(0);
#else
  asm volatile("s_wait_asynccnt 0" ::: "memory");
#endif
#endif
}

// ---------------------------------------------------------------------------
// Kernel 0a: W[n,d,m,v] fp32 -> bf16 hi/lo planes, transposed to [n][col][d]
// ---------------------------------------------------------------------------
__global__ __launch_bounds__(256)
void split_w_kernel(const float* __restrict__ W,
                    unsigned short* __restrict__ hi, unsigned short* __restrict__ lo) {
  int t = blockIdx.x * 256 + threadIdx.x;        // 358,400 tasks
  int col = t % MV;
  int r = t / MV;                                 // 0..223
  int d0 = (r & 31) * 8;
  int n = r >> 5;
  unsigned short h[8], l[8];
#pragma unroll
  for (int j = 0; j < 8; ++j) {
    float x = W[((size_t)(n * DDIM + d0 + j) * MV) + col];   // coalesced over col
    h[j] = f2bf(x);
    l[j] = f2bf(x - bf2f(h[j]));
  }
  size_t o = ((size_t)n * MV + col) * DDIM + d0;
  pack8((unsigned short*)&hi[o], h);
  pack8((unsigned short*)&lo[o], l);
}

// ---------------------------------------------------------------------------
// Kernel 0b: prim_pose[b,n,d] fp32 -> bf16 hi/lo planes, transposed to [n][b][d]
// ---------------------------------------------------------------------------
__global__ __launch_bounds__(256)
void split_a_kernel(const float* __restrict__ P,
                    unsigned short* __restrict__ hi, unsigned short* __restrict__ lo) {
  int t = blockIdx.x * 256 + threadIdx.x;        // 1,835,008 tasks
  int d0 = (t & 31) * 8;
  int bn = t >> 5;                                // b*NCAP + n
  int n = bn % NCAP;
  int b = bn / NCAP;
  const float* g = P + (size_t)bn * DDIM + d0;    // contiguous across t
  float4 f0 = *(const float4*)(g);
  float4 f1 = *(const float4*)(g + 4);
  float x[8] = {f0.x, f0.y, f0.z, f0.w, f1.x, f1.y, f1.z, f1.w};
  unsigned short h[8], l[8];
#pragma unroll
  for (int j = 0; j < 8; ++j) {
    h[j] = f2bf(x[j]);
    l[j] = f2bf(x[j] - bf2f(h[j]));
  }
  size_t o = ((size_t)n * BTOT + b) * DDIM + d0;
  pack8((unsigned short*)&hi[o], h);
  pack8((unsigned short*)&lo[o], l);
}

// ---------------------------------------------------------------------------
// Kernel 1: votes[b,n,mv] (bf16) = pose[b,n,:] @ W[n,:,mv], bf16 hi/lo split.
// WG = 8 waves (2 row-groups x 4 col-groups), wave tile 32b x 80mv:
// per k-step 30 wmma vs 28 ds_load_b128. Double-buffered LDS + async prefetch.
// ---------------------------------------------------------------------------
#define KPAD  40                          // row stride (bf16): 80B, 16B-aligned
#define BT    64
#define CT    320
#define NSTEP (DDIM / 32)
#define AS_PLANE_B (BT * KPAD * 2)        // 5120
#define AS_BUF_B   (2 * AS_PLANE_B)       // 10240
#define BS_PLANE_B (CT * KPAD * 2)        // 25600
#define BS_BUF_B   (2 * BS_PLANE_B)       // 51200
#define BS_OFF     (2 * AS_BUF_B)         // 20480
#define GEMM_SMEM  (BS_OFF + 2 * BS_BUF_B) // 122880

__device__ __forceinline__ void gemm_stage(int step, int buf, unsigned char* smem,
                                           const unsigned short* Ahi, const unsigned short* Alo,
                                           const unsigned short* Whi, const unsigned short* Wlo,
                                           int tid, int b0, int c0, int n) {
  const int k0 = step * 32;
  { // A tile: 64 rows x 32 k, hi+lo (2 x 16B per thread)
    int kk = tid & 3, row = (tid >> 2) & 63;
    size_t goff = ((size_t)(n * BTOT + b0 + row) * DDIM) + k0 + kk * 8;
    unsigned loff = (unsigned)buf * AS_BUF_B + (unsigned)(row * KPAD + kk * 8) * 2;
    cp16(Ahi + goff, smem + loff);
    cp16(Alo + goff, smem + AS_PLANE_B + loff);
  }
  // B tile: 320 cols x 32 k, hi+lo (10 x 16B per thread)
#pragma unroll
  for (int i = 0; i < 5; ++i) {
    int tau = tid + 256 * i;              // 0..1279
    int kk = tau & 3, col = tau >> 2;     // col 0..319
    size_t goff = ((size_t)(n * MV + c0 + col) * DDIM) + k0 + kk * 8;
    unsigned loff = BS_OFF + (unsigned)buf * BS_BUF_B + (unsigned)(col * KPAD + kk * 8) * 2;
    cp16(Whi + goff, smem + loff);
    cp16(Wlo + goff, smem + BS_PLANE_B + loff);
  }
}

__global__ __launch_bounds__(256)
void votes_gemm_kernel(const unsigned short* __restrict__ Ahi, const unsigned short* __restrict__ Alo,
                       const unsigned short* __restrict__ Whi, const unsigned short* __restrict__ Wlo,
                       unsigned short* __restrict__ votes) {
  extern __shared__ unsigned char smem[];

  const int b0   = blockIdx.x * BT;
  const int c0   = blockIdx.y * CT;
  const int n    = blockIdx.z;
  const int tid  = threadIdx.x;
  const int lane = tid & 31;
  const int wave = tid >> 5;
  const int rg   = wave >> 2;     // 0..1 : 32-row group
  const int cg   = wave & 3;      // 0..3 : 80-col group
  const int lpos = lane & 15;
  const int half = lane >> 4;

  v8f acc[2][5];
#pragma unroll
  for (int rt = 0; rt < 2; ++rt)
#pragma unroll
    for (int ct = 0; ct < 5; ++ct)
#pragma unroll
      for (int e = 0; e < 8; ++e) acc[rt][ct][e] = 0.f;

  gemm_stage(0, 0, smem, Ahi, Alo, Whi, Wlo, tid, b0, c0, n);

  for (int step = 0; step < NSTEP; ++step) {
    const int buf = step & 1;
    wait_async0();                 // own async chunks for buf[cur] landed
    __syncthreads();               // everyone's staging visible; prev reads done
    if (step + 1 < NSTEP)
      gemm_stage(step + 1, buf ^ 1, smem, Ahi, Alo, Whi, Wlo, tid, b0, c0, n);

    union Frag { U4 q[2]; v16bf v; };
    Frag fahi[2], falo[2];
#pragma unroll
    for (int rt = 0; rt < 2; ++rt) {
      // A 16x32 bf16: lanes 0-15 row M=lpos, K{0-7,16-23}; lanes 16-31 K{8-15,24-31}
      const unsigned short* ah = (const unsigned short*)(smem + buf * AS_BUF_B) + (rg * 32 + rt * 16 + lpos) * KPAD;
      const unsigned short* al = ah + BT * KPAD;   // lo plane
      fahi[rt].q[0] = *(const U4*)(ah + half * 8);
      fahi[rt].q[1] = *(const U4*)(ah + 16 + half * 8);
      falo[rt].q[0] = *(const U4*)(al + half * 8);
      falo[rt].q[1] = *(const U4*)(al + 16 + half * 8);
    }
#pragma unroll
    for (int ct = 0; ct < 5; ++ct) {
      // B 32x16 bf16: lanes 0-15 col N=lpos K{0-7,16-23}; lanes 16-31 K{8-15,24-31}
      const unsigned short* bh = (const unsigned short*)(smem + BS_OFF + buf * BS_BUF_B) + (cg * 80 + ct * 16 + lpos) * KPAD;
      const unsigned short* bl = bh + CT * KPAD;   // lo plane
      Frag fbhi, fblo;
      fbhi.q[0] = *(const U4*)(bh + half * 8);
      fbhi.q[1] = *(const U4*)(bh + 16 + half * 8);
      fblo.q[0] = *(const U4*)(bl + half * 8);
      fblo.q[1] = *(const U4*)(bl + 16 + half * 8);
#pragma unroll
      for (int rt = 0; rt < 2; ++rt) {
        acc[rt][ct] = __builtin_amdgcn_wmma_f32_16x16x32_bf16(false, fahi[rt].v, false, fbhi.v,
                                                              (short)0, acc[rt][ct], false, false);
        acc[rt][ct] = __builtin_amdgcn_wmma_f32_16x16x32_bf16(false, fahi[rt].v, false, fblo.v,
                                                              (short)0, acc[rt][ct], false, false);
        acc[rt][ct] = __builtin_amdgcn_wmma_f32_16x16x32_bf16(false, falo[rt].v, false, fbhi.v,
                                                              (short)0, acc[rt][ct], false, false);
      }
    }
  }

  // epilogue: C layout VGPR j -> row M=j (+8 for lanes 16-31), N=lpos
#pragma unroll
  for (int rt = 0; rt < 2; ++rt)
#pragma unroll
    for (int ct = 0; ct < 5; ++ct) {
      int col = c0 + cg * 80 + ct * 16 + lpos;
#pragma unroll
      for (int j = 0; j < 8; ++j) {
        int b = b0 + rg * 32 + rt * 16 + half * 8 + j;
        votes[((size_t)b * NCAP + n) * MV + col] = f2bf(acc[rt][ct][j]);
      }
    }
}

// ---------------------------------------------------------------------------
// Kernel 2: fused routing + layernorm + softmax + logits.
// 1 WG = 8 batch rows; votes tile (179.2 KB bf16) async-copied to LDS once.
// All dot-reductions 8-lane cooperative (conflict-free b128 LDS reads + shfl).
// ---------------------------------------------------------------------------
#define B2 8
#define V_BYTES (B2 * NCAP * MV * 2)              // 179200
#define P_BYTES (B2 * MCAP * VDIM * 4)            // 51200
#define C_BYTES (B2 * NCAP * MCAP * 4)            // 5600
#define A_BYTES (B2 * NCAP * 4)                   // 224
#define SMEM2   (V_BYTES + P_BYTES + C_BYTES + A_BYTES)

__device__ __forceinline__ void ln_rows_coop(float* pose, const float* g, const float* be,
                                             int sub, int rbase) {
  float4 g0 = *(const float4*)&g[sub * 8];
  float4 g1 = *(const float4*)&g[sub * 8 + 4];
  float4 e0 = *(const float4*)&be[sub * 8];
  float4 e1 = *(const float4*)&be[sub * 8 + 4];
  for (int row = rbase; row < B2 * MCAP; row += 32) {
    float* p = &pose[row * VDIM + sub * 8];
    float4 a = ((float4*)p)[0];
    float4 c = ((float4*)p)[1];
    float s = a.x + a.y + a.z + a.w + c.x + c.y + c.z + c.w;
    float mu = shfl_red8(s) * (1.0f / VDIM);
    float d, vs = 0.f;
    d = a.x - mu; vs += d * d;  d = a.y - mu; vs += d * d;
    d = a.z - mu; vs += d * d;  d = a.w - mu; vs += d * d;
    d = c.x - mu; vs += d * d;  d = c.y - mu; vs += d * d;
    d = c.z - mu; vs += d * d;  d = c.w - mu; vs += d * d;
    float inv = rsqrtf(shfl_red8(vs) * (1.0f / VDIM) + 1e-5f);
    a.x = (a.x - mu) * inv * g0.x + e0.x;  a.y = (a.y - mu) * inv * g0.y + e0.y;
    a.z = (a.z - mu) * inv * g0.z + e0.z;  a.w = (a.w - mu) * inv * g0.w + e0.w;
    c.x = (c.x - mu) * inv * g1.x + e1.x;  c.y = (c.y - mu) * inv * g1.y + e1.y;
    c.z = (c.z - mu) * inv * g1.z + e1.z;  c.w = (c.w - mu) * inv * g1.w + e1.w;
    ((float4*)p)[0] = a;
    ((float4*)p)[1] = c;
  }
}

__global__ __launch_bounds__(256)
void routing_fused_kernel(const unsigned short* __restrict__ votes,
                          const float* __restrict__ prim_act,
                          const float* __restrict__ ln_g,
                          const float* __restrict__ ln_b,
                          const float* __restrict__ emb,
                          const float* __restrict__ bias,
                          const int* __restrict__ nr_p,
                          float* __restrict__ out) {
  extern __shared__ unsigned char smem[];
  unsigned short* vl   = (unsigned short*)smem;
  float*          pose = (float*)(smem + V_BYTES);
  float*          cbuf = (float*)(smem + V_BYTES + P_BYTES);
  float*          actl = (float*)(smem + V_BYTES + P_BYTES + C_BYTES);

  const int tid  = threadIdx.x;
  const int b0   = blockIdx.x * B2;
  const int nr   = *nr_p;
  const int sub  = tid & 7;        // lane-in-group for cooperative reductions
  const int rbse = tid >> 3;       // group id 0..31

  // async bulk copy of the votes tile (11200 x 16B)
  {
    const unsigned char* g = (const unsigned char*)(votes + (size_t)b0 * NCAP * MV);
    for (int r = tid; r < V_BYTES / 16; r += 256)
      cp16(g + (size_t)r * 16, smem + (size_t)r * 16);
  }
  for (int r = tid; r < B2 * NCAP; r += 256) {
    float a = prim_act[(size_t)b0 * NCAP + r];
    actl[r] = a;
    out[(size_t)BTOT * MCAP + (size_t)b0 * NCAP + r] = a;   // prim_act passthrough
  }
  for (int r = tid; r < B2 * NCAP * MCAP; r += 256) cbuf[r] = 1.0f / MCAP;
  wait_async0();
  __syncthreads();

  // iteration 0: pose = sum_n (act/M) * votes   (8-wide chunks, lane-contiguous)
  for (int r = tid; r < B2 * MCAP * VDIM / 8; r += 256) {   // 1600 chunks
    int b = r / 200;
    int mv0 = (r % 200) * 8;
    float s0 = 0, s1 = 0, s2 = 0, s3 = 0, s4 = 0, s5 = 0, s6 = 0, s7 = 0;
#pragma unroll
    for (int nn = 0; nn < NCAP; ++nn) {
      float a = actl[b * NCAP + nn];
      U4 q = *(const U4*)&vl[(b * NCAP + nn) * MV + mv0];
      s0 += a * lo16f(q.x); s1 += a * hi16f(q.x);
      s2 += a * lo16f(q.y); s3 += a * hi16f(q.y);
      s4 += a * lo16f(q.z); s5 += a * hi16f(q.z);
      s6 += a * lo16f(q.w); s7 += a * hi16f(q.w);
    }
    const float k = 1.0f / MCAP;
    float4 o0, o1;
    o0.x = s0 * k; o0.y = s1 * k; o0.z = s2 * k; o0.w = s3 * k;
    o1.x = s4 * k; o1.y = s5 * k; o1.z = s6 * k; o1.w = s7 * k;
    ((float4*)&pose[r * 8])[0] = o0;
    ((float4*)&pose[r * 8 + 4])[0] = o1;
  }
  __syncthreads();
  ln_rows_coop(pose, ln_g, ln_b, sub, rbse);
  __syncthreads();

  const float scale = 0.125f;   // 1/sqrt(V)
  for (int it = 1; it < nr; ++it) {
    // qk[b,n,m]: 8-lane cooperative dots over v=64
    for (int dt = rbse; dt < B2 * NCAP * MCAP; dt += 32) {
      int b = dt / (NCAP * MCAP);
      int nm = dt % (NCAP * MCAP);
      int nn = nm / MCAP, m = nm % MCAP;
      U4 q = *(const U4*)&vl[(b * NCAP + nn) * MV + m * VDIM + sub * 8];
      const float4* pq = (const float4*)&pose[(b * MCAP + m) * VDIM + sub * 8];
      float4 f0 = pq[0], f1 = pq[1];
      float s = lo16f(q.x) * f0.x + hi16f(q.x) * f0.y + lo16f(q.y) * f0.z + hi16f(q.y) * f0.w
              + lo16f(q.z) * f1.x + hi16f(q.z) * f1.y + lo16f(q.w) * f1.z + hi16f(q.w) * f1.w;
      s = shfl_red8(s);
      if (sub == 0) cbuf[dt] = s * scale;
    }
    __syncthreads();
    for (int r = tid; r < B2 * NCAP; r += 256) {       // softmax over m
      float* row = &cbuf[r * MCAP];
      float mx = row[0];
      for (int m = 1; m < MCAP; ++m) mx = fmaxf(mx, row[m]);
      float sum = 0.f;
      for (int m = 0; m < MCAP; ++m) { float e = __expf(row[m] - mx); row[m] = e; sum += e; }
      float inv = 1.0f / sum;
      for (int m = 0; m < MCAP; ++m) row[m] *= inv;
    }
    __syncthreads();
    // pose[b,m,v] = sum_n c[b,n,m]*act[b,n]*votes[b,n,m,v]  (8-wide chunks)
    for (int r = tid; r < B2 * MCAP * VDIM / 8; r += 256) {
      int b = r / 200;
      int mv0 = (r % 200) * 8;
      int m = mv0 >> 6;
      float s0 = 0, s1 = 0, s2 = 0, s3 = 0, s4 = 0, s5 = 0, s6 = 0, s7 = 0;
#pragma unroll
      for (int nn = 0; nn < NCAP; ++nn) {
        float cw = cbuf[(b * NCAP + nn) * MCAP + m] * actl[b * NCAP + nn];
        U4 q = *(const U4*)&vl[(b * NCAP + nn) * MV + mv0];
        s0 += cw * lo16f(q.x); s1 += cw * hi16f(q.x);
        s2 += cw * lo16f(q.y); s3 += cw * hi16f(q.y);
        s4 += cw * lo16f(q.z); s5 += cw * hi16f(q.z);
        s6 += cw * lo16f(q.w); s7 += cw * hi16f(q.w);
      }
      float4 o0, o1;
      o0.x = s0; o0.y = s1; o0.z = s2; o0.w = s3;
      o1.x = s4; o1.y = s5; o1.z = s6; o1.w = s7;
      ((float4*)&pose[r * 8])[0] = o0;
      ((float4*)&pose[r * 8 + 4])[0] = o1;
    }
    __syncthreads();
    ln_rows_coop(pose, ln_g, ln_b, sub, rbse);
    __syncthreads();
  }

  // logits: 8-lane cooperative dots of pose[b,m,:] with embedding[m,:]
  for (int dt = rbse; dt < B2 * MCAP; dt += 32) {
    int b = dt / MCAP, m = dt % MCAP;
    const float4* pq = (const float4*)&pose[dt * VDIM + sub * 8];
    const float4* eq = (const float4*)&emb[m * VDIM + sub * 8];
    float4 p0 = pq[0], p1 = pq[1], e0 = eq[0], e1 = eq[1];
    float s = p0.x * e0.x + p0.y * e0.y + p0.z * e0.z + p0.w * e0.w
            + p1.x * e1.x + p1.y * e1.y + p1.z * e1.z + p1.w * e1.w;
    s = shfl_red8(s);
    if (sub == 0) out[(size_t)(b0 + b) * MCAP + m] = s + bias[m];
  }
  for (int r = tid; r < B2 * NCAP * MCAP; r += 256)     // final c
    out[(size_t)BTOT * MCAP + (size_t)BTOT * NCAP + (size_t)b0 * NCAP * MCAP + r] = cbuf[r];
}

// ---------------------------------------------------------------------------
extern "C" void kernel_launch(void* const* d_in, const int* in_sizes, int n_in,
                              void* d_out, int out_size, void* d_ws, size_t ws_size,
                              hipStream_t stream) {
  (void)in_sizes; (void)n_in; (void)out_size; (void)ws_size;
  const float* prim_pose = (const float*)d_in[0];
  const float* prim_act  = (const float*)d_in[1];
  const float* W         = (const float*)d_in[2];
  const float* ln_g      = (const float*)d_in[3];
  const float* ln_b      = (const float*)d_in[4];
  const float* emb       = (const float*)d_in[5];
  const float* bias      = (const float*)d_in[6];
  const int*   nr        = (const int*)d_in[7];
  float* out = (float*)d_out;

  unsigned char* ws = (unsigned char*)d_ws;
  unsigned short* votes = (unsigned short*)ws;
  unsigned short* Whi = (unsigned short*)(ws + WHI_OFF);
  unsigned short* Wlo = (unsigned short*)(ws + WLO_OFF);
  unsigned short* Ahi = (unsigned short*)(ws + AHI_OFF);
  unsigned short* Alo = (unsigned short*)(ws + ALO_OFF);

  split_w_kernel<<<dim3(1400), 256, 0, stream>>>(W, Whi, Wlo);
  split_a_kernel<<<dim3(7168), 256, 0, stream>>>(prim_pose, Ahi, Alo);

  (void)hipFuncSetAttribute(reinterpret_cast<const void*>(votes_gemm_kernel),
                            hipFuncAttributeMaxDynamicSharedMemorySize, (int)GEMM_SMEM);
  dim3 g1(BTOT / BT, MV / CT, NCAP);               // 128 x 5 x 7
  votes_gemm_kernel<<<g1, 256, GEMM_SMEM, stream>>>(Ahi, Alo, Whi, Wlo, votes);

  (void)hipFuncSetAttribute(reinterpret_cast<const void*>(routing_fused_kernel),
                            hipFuncAttributeMaxDynamicSharedMemorySize, (int)SMEM2);
  routing_fused_kernel<<<dim3(BTOT / B2), 256, SMEM2, stream>>>(
      votes, prim_act, ln_g, ln_b, emb, bias, nr, out);
}
